// RelationalCrossModal_14010183320211
// MI455X (gfx1250) — matrix operand order repaired and checked by reading, not compiled, not run on hardware
//
#include <hip/hip_runtime.h>
#include <hip/hip_bf16.h>

// ---- CDNA5 / gfx1250: wave32, WMMA 16x16x32 f16 -> f32 ----
typedef __attribute__((ext_vector_type(16))) _Float16 v16h;
typedef __attribute__((ext_vector_type(8)))  _Float16 v8h;
typedef __attribute__((ext_vector_type(8)))  float    v8f;

#define B_  4
#define N1_ 128
#define N2_ 128
#define D_  256
#define R_  4

__device__ __forceinline__ v8f wmma16x16x32(v16h a, v16h b, v8f c) {
  // D = A(16x32 f16) x B(32x16 f16) + C(16x16 f32)
  return __builtin_amdgcn_wmma_f32_16x16x32_f16(false, a, false, b, (short)0, c,
                                                false, false);
}

// A-fragment (row-major, ld = ldk halves), row pointer supplied by caller.
// lane: h = lane>>4. elems 0..15 -> K = kbase+16h+e
__device__ __forceinline__ v16h frag_a_row(const _Float16* row, int kbase, int h) {
  const _Float16* p = row + kbase + 16 * h;
  v8h lo = *(const v8h*)(p);
  v8h hi = *(const v8h*)(p + 8);
  v16h r;
#pragma unroll
  for (int e = 0; e < 8; ++e) { r[e] = lo[e]; r[e + 8] = hi[e]; }
  return r;
}

// B-fragment from pre-transposed weights Wt[n][k], n folded into pointer.
// lane: K = kbase + 16h + e (contiguous 16 halves = 32 B)
__device__ __forceinline__ v16h frag_b(const _Float16* rowN, int kbase, int h) {
  return *(const v16h*)(rowN + kbase + 16 * h);
}

__device__ __forceinline__ v16h relu16(v16h x) {
  v16h z = {};
#if __has_builtin(__builtin_elementwise_max)
  return __builtin_elementwise_max(x, z);   // v_pk_max_num_f16
#else
  v16h r;
#pragma unroll
  for (int e = 0; e < 16; ++e) r[e] = x[e] > z[e] ? x[e] : z[e];
  return r;
#endif
}

// ---------------- prep kernels ----------------
__global__ void k_cvt_f16(const float* __restrict__ src, _Float16* __restrict__ dst,
                          int n) {
  int i = blockIdx.x * blockDim.x + threadIdx.x;
  if (i < n) dst[i] = (_Float16)src[i];
}

// dst[bt][c][r] = (f16) src[bt*src_bs + (row_off + r)*cols + c]
__global__ void k_transpose_f16(const float* __restrict__ src,
                                _Float16* __restrict__ dst, int batch, int rows,
                                int cols, long src_bs, int row_off) {
  long total = (long)batch * rows * cols;
  long i = (long)blockIdx.x * blockDim.x + threadIdx.x;
  if (i >= total) return;
  int c = (int)(i % cols);
  long t = i / cols;
  int r = (int)(t % rows);
  int bt = (int)(t / rows);
  dst[(long)bt * rows * cols + (long)c * rows + r] =
      (_Float16)src[(long)bt * src_bs + (long)(row_off + r) * cols + c];
}

// ---------------- stage 1: H[b][r][i][d] = f16(F[b] @ Wt[r]^T (+ bias)) ----
__global__ void __launch_bounds__(128)
k_stage1(const _Float16* __restrict__ F, const _Float16* __restrict__ Wt,
         const float* __restrict__ bias, _Float16* __restrict__ H) {
  int wave = threadIdx.x >> 5;
  int lane = threadIdx.x & 31;
  int t = blockIdx.x * 4 + wave;      // 2048 tiles total
  int dt = t & 15; t >>= 4;           // 16 d-tiles
  int it = t & 7;  t >>= 3;           // 8  i-tiles
  int r  = t & 3;  t >>= 2;           // R
  int b  = t;                         // B
  int m = lane & 15, h = lane >> 4;

  const _Float16* Arows = F + ((long)b * N1_ + it * 16) * D_;
  const _Float16* Brow  = Wt + (long)r * D_ * D_ + (long)(dt * 16 + m) * D_;

  v8f acc = {};
#pragma unroll
  for (int ks = 0; ks < 8; ++ks)
    acc = wmma16x16x32(frag_a_row(Arows + (long)m * D_, ks * 32, h),
                       frag_b(Brow, ks * 32, h), acc);

  int dg = dt * 16 + m;
  float bv = bias ? bias[r * D_ + dg] : 0.0f;
  _Float16* Hrow = H + (((long)b * R_ + r) * N1_ + it * 16) * D_;
#pragma unroll
  for (int v = 0; v < 8; ++v)
    Hrow[(long)(v + 8 * h) * D_ + dg] = (_Float16)(acc[v] + bv);
}

// ---------------- stage 2: fused pairwise relu + GEMM + softmax + rel ------
// grid: B*R*(N1/16) = 128 blocks, 256 threads (8 waves, one 16-j subtile each)
// dynamic LDS (halves):
//   h1 16x256 (8K) | h2 128x256 (64K) | w2 128x256 (64K) | sc 16x128 f32 (8K)
//   | w 16x128 (4K)  == 148 KB  (CDNA5 WGP has 320 KB; beyond 64K static cap)
#define LDS_H1_OFF 0
#define LDS_H2_OFF (16 * 256)
#define LDS_W2_OFF (LDS_H2_OFF + 128 * 256)
#define LDS_SC_OFF (LDS_W2_OFF + 128 * 256)
#define LDS_W_OFF  (LDS_SC_OFF + 16 * 128 * 2)       // sc is f32 = 2 halves
#define LDS_TOTAL_BYTES ((LDS_W_OFF + 16 * 128) * 2)

__global__ void __launch_bounds__(256)
k_pairs(const _Float16* __restrict__ H1, const _Float16* __restrict__ H2,
        const _Float16* __restrict__ W2t, const float* __restrict__ b2g,
        const float* __restrict__ w3g, const _Float16* __restrict__ F2t,
        _Float16* __restrict__ AR) {
  extern __shared__ _Float16 smem[];
  _Float16* sm_h1 = smem + LDS_H1_OFF;
  _Float16* sm_h2 = smem + LDS_H2_OFF;
  _Float16* sm_w2 = smem + LDS_W2_OFF;
  float*    sm_sc = (float*)(smem + LDS_SC_OFF);
  _Float16* sm_w  = smem + LDS_W_OFF;

  int t = blockIdx.x;                  // 128 = B*R*8
  int it = t & 7; t >>= 3;
  int r  = t & 3; t >>= 2;
  int b  = t;
  int wave = threadIdx.x >> 5, lane = threadIdx.x & 31;
  int m = lane & 15, h = lane >> 4;
  int i0 = it * 16;
  int j0 = wave * 16;                  // this wave's 16 j's

  const _Float16* W2r = W2t + (long)r * 128 * D_;
  __builtin_prefetch(W2r + (long)threadIdx.x * 128, 0, 3);  // global_prefetch_b8

  // cooperative LDS fill: H1 tile (2048 dw) + H2 tile + W2 tile (16384 dw each)
  {
    const unsigned* s1 = (const unsigned*)(H1 + (((long)b * R_ + r) * N1_ + i0) * D_);
    const unsigned* s2 = (const unsigned*)(H2 + ((long)b * R_ + r) * N2_ * D_);
    const unsigned* s3 = (const unsigned*)W2r;
    unsigned* d1 = (unsigned*)sm_h1;
    unsigned* d2 = (unsigned*)sm_h2;
    unsigned* d3 = (unsigned*)sm_w2;
    for (int c = threadIdx.x; c < 2048; c += 256) d1[c] = s1[c];
    for (int c = threadIdx.x; c < 16384; c += 256) d2[c] = s2[c];
    for (int c = threadIdx.x; c < 16384; c += 256) d3[c] = s3[c];
  }
  __syncthreads();

  // per-lane loop invariants: this lane's output column ng = nt*16 + m
  float b2v[8], w3v[8];
#pragma unroll
  for (int nt = 0; nt < 8; ++nt) {
    b2v[nt] = b2g[r * 128 + nt * 16 + m];
    w3v[nt] = w3g[r * 128 + nt * 16 + m];
  }

  // H2 A-fragment halves: invariant over i -> keep in registers
  v16h h2f[8];
#pragma unroll
  for (int ks = 0; ks < 8; ++ks)
    h2f[ks] = frag_a_row(sm_h2 + (long)(j0 + m) * D_, ks * 32, h);

  for (int il = 0; il < 16; ++il) {
    // afr = relu(h1[il] + h2[j0+m]) assembled directly in registers
    const _Float16* h1row = sm_h1 + (long)il * D_;
    v16h afr[8];
#pragma unroll
    for (int ks = 0; ks < 8; ++ks)
      afr[ks] = relu16(frag_a_row(h1row, ks * 32, h) + h2f[ks]);  // pk_add/pk_max

    float part[8];
#pragma unroll
    for (int v = 0; v < 8; ++v) part[v] = 0.0f;

    for (int nt = 0; nt < 8; ++nt) {
      const _Float16* Brow = sm_w2 + (long)(nt * 16 + m) * D_;   // LDS-resident W2
      v8f acc = {};
#pragma unroll
      for (int ks = 0; ks < 8; ++ks)
        acc = wmma16x16x32(afr[ks], frag_b(Brow, ks * 32, h), acc);
#pragma unroll
      for (int v = 0; v < 8; ++v) {
        float g = acc[v] + b2v[nt];
        g = g > 0.0f ? g : 0.0f;
        part[v] += g * w3v[nt];        // score contribution of this lane's column
      }
    }
    // sum over the 16 columns held by the 16 lanes of each half-group
#pragma unroll
    for (int v = 0; v < 8; ++v) {
      float s = part[v];
      s += __shfl_xor(s, 1, 32);
      s += __shfl_xor(s, 2, 32);
      s += __shfl_xor(s, 4, 32);
      s += __shfl_xor(s, 8, 32);
      part[v] = s;
    }
    if (m == 0) {                      // lanes 0 / 16 write M=0..7 / 8..15
#pragma unroll
      for (int v = 0; v < 8; ++v)
        sm_sc[il * 128 + j0 + v + 8 * h] = part[v];
    }
  }
  __syncthreads();

  // softmax over j (b3 constant per row -> cancels); 2 rows per wave
  for (int q = 0; q < 2; ++q) {
    int il = wave * 2 + q;
    float x[4];
#pragma unroll
    for (int tt = 0; tt < 4; ++tt) x[tt] = sm_sc[il * 128 + tt * 32 + lane];
    float mx = fmaxf(fmaxf(x[0], x[1]), fmaxf(x[2], x[3]));
    for (int msk = 1; msk < 32; msk <<= 1) mx = fmaxf(mx, __shfl_xor(mx, msk, 32));
    float sm = 0.0f;
#pragma unroll
    for (int tt = 0; tt < 4; ++tt) { x[tt] = __expf(x[tt] - mx); sm += x[tt]; }
    for (int msk = 1; msk < 32; msk <<= 1) sm += __shfl_xor(sm, msk, 32);
    float inv = 1.0f / sm;
#pragma unroll
    for (int tt = 0; tt < 4; ++tt)
      sm_w[il * 128 + tt * 32 + lane] = (_Float16)(x[tt] * inv);
  }
  __syncthreads();

  // rel = w(16x128) @ F2^T -> 16x256 ; 2 d-tiles per wave, WMMA again
  const _Float16* F2b = F2t + (long)b * D_ * N2_;
  for (int q = 0; q < 2; ++q) {
    int nt = wave * 2 + q;             // 0..15
    int ng = nt * 16 + m;
    const _Float16* Brow = F2b + (long)ng * N2_;
    v8f acc = {};
#pragma unroll
    for (int ks = 0; ks < 4; ++ks)
      acc = wmma16x16x32(frag_a_row(sm_w + (long)m * 128, ks * 32, h),
                         frag_b(Brow, ks * 32, h), acc);
#pragma unroll
    for (int v = 0; v < 8; ++v) {
      int M = v + 8 * h;
      AR[((long)b * N1_ + i0 + M) * (R_ * D_) + r * D_ + ng] = (_Float16)acc[v];
    }
  }
}

// ---------------- stage 3: out = relu(AR @ Wa1 + ba1) @ Wa2 + ba2 ----------
__global__ void __launch_bounds__(256)
k_mlp(const _Float16* __restrict__ AR, const _Float16* __restrict__ Wa1t,
      const float* __restrict__ ba1, const _Float16* __restrict__ Wa2t,
      const float* __restrict__ ba2, float* __restrict__ out) {
  __shared__ alignas(32) _Float16 lds_ar[16 * 1024];   // 32 KB
  __shared__ alignas(32) _Float16 lds_t[16 * 512];     // 16 KB
  int t = blockIdx.x;                  // 32 = B*8
  int it = t & 7;
  int b  = t >> 3;
  int i0 = it * 16;
  int wave = threadIdx.x >> 5, lane = threadIdx.x & 31;
  int m = lane & 15, h = lane >> 4;

  {
    const unsigned* s = (const unsigned*)(AR + ((long)b * N1_ + i0) * 1024);
    unsigned* d = (unsigned*)lds_ar;
    for (int c = threadIdx.x; c < 8192; c += 256) d[c] = s[c];
  }
  __syncthreads();

  // GEMM1: 16x1024 @ (512x1024)^T -> relu+ba1 -> lds_t (f16)
  for (int q = 0; q < 4; ++q) {
    int nt = wave * 4 + q;             // 0..31
    int ng = nt * 16 + m;
    const _Float16* Brow = Wa1t + (long)ng * 1024;
    v8f acc = {};
    for (int ks = 0; ks < 32; ++ks)
      acc = wmma16x16x32(frag_a_row(lds_ar + (long)m * 1024, ks * 32, h),
                         frag_b(Brow, ks * 32, h), acc);
    float bv = ba1[ng];
#pragma unroll
    for (int v = 0; v < 8; ++v) {
      float g = acc[v] + bv;
      g = g > 0.0f ? g : 0.0f;
      lds_t[(long)(v + 8 * h) * 512 + ng] = (_Float16)g;
    }
  }
  __syncthreads();

  // GEMM2: 16x512 @ (256x512)^T + ba2 -> out (f32)
  for (int q = 0; q < 2; ++q) {
    int nt = wave * 2 + q;             // 0..15
    int ng = nt * 16 + m;
    const _Float16* Brow = Wa2t + (long)ng * 512;
    v8f acc = {};
    for (int ks = 0; ks < 16; ++ks)
      acc = wmma16x16x32(frag_a_row(lds_t + (long)m * 512, ks * 32, h),
                         frag_b(Brow, ks * 32, h), acc);
    float bv = ba2[ng];
#pragma unroll
    for (int v = 0; v < 8; ++v)
      out[((long)b * N1_ + i0 + v + 8 * h) * 256 + ng] = acc[v] + bv;
  }
}

// ---------------- launch ----------------
extern "C" void kernel_launch(void* const* d_in, const int* in_sizes, int n_in,
                              void* d_out, int out_size, void* d_ws, size_t ws_size,
                              hipStream_t stream) {
  const float* f1  = (const float*)d_in[0];
  const float* f2  = (const float*)d_in[1];
  const float* W1  = (const float*)d_in[2];
  const float* b1  = (const float*)d_in[3];
  const float* W2  = (const float*)d_in[4];
  const float* b2  = (const float*)d_in[5];
  const float* w3  = (const float*)d_in[6];
  // d_in[7] = b3: constant per softmax row -> cancels, unused
  const float* Wa1 = (const float*)d_in[8];
  const float* ba1 = (const float*)d_in[9];
  const float* Wa2 = (const float*)d_in[10];
  const float* ba2 = (const float*)d_in[11];

  char* ws = (char*)d_ws;
  _Float16* F1h  = (_Float16*)(ws + 0);        // 256 KB
  _Float16* F2h  = (_Float16*)(ws + 262144);   // 256 KB
  _Float16* F2t  = (_Float16*)(ws + 524288);   // 256 KB  [b][d][j]
  _Float16* W1at = (_Float16*)(ws + 786432);   // 512 KB  [r][e][k]
  _Float16* W1bt = (_Float16*)(ws + 1310720);  // 512 KB
  _Float16* W2t  = (_Float16*)(ws + 1835008);  // 256 KB  [r][n][k]
  _Float16* Wa1t = (_Float16*)(ws + 2097152);  // 1 MB    [n][k]
  _Float16* Wa2t = (_Float16*)(ws + 3145728);  // 256 KB
  _Float16* H1   = (_Float16*)(ws + 3407872);  // 1 MB    [b][r][i][d]
  _Float16* H2   = (_Float16*)(ws + 4456448);  // 1 MB
  _Float16* AR   = (_Float16*)(ws + 5505024);  // 1 MB    [b][i][r*D+d]

  k_cvt_f16<<<512, 256, 0, stream>>>(f1, F1h, B_ * N1_ * D_);
  k_cvt_f16<<<512, 256, 0, stream>>>(f2, F2h, B_ * N2_ * D_);
  k_transpose_f16<<<512,  256, 0, stream>>>(f2,  F2t,  B_, N2_, D_,  (long)N2_ * D_, 0);
  k_transpose_f16<<<1024, 256, 0, stream>>>(W1,  W1at, R_, D_,  D_,  (long)2 * D_ * D_, 0);
  k_transpose_f16<<<1024, 256, 0, stream>>>(W1,  W1bt, R_, D_,  D_,  (long)2 * D_ * D_, D_);
  k_transpose_f16<<<512,  256, 0, stream>>>(W2,  W2t,  R_, D_,  128, (long)D_ * 128, 0);
  k_transpose_f16<<<2048, 256, 0, stream>>>(Wa1, Wa1t, 1, 1024, 512, (long)1024 * 512, 0);
  k_transpose_f16<<<512,  256, 0, stream>>>(Wa2, Wa2t, 1, 512,  256, (long)512 * 256, 0);

  k_stage1<<<512, 128, 0, stream>>>(F1h, W1at, b1, H1);        // h1 + b1
  k_stage1<<<512, 128, 0, stream>>>(F2h, W1bt, nullptr, H2);   // h2

  k_pairs<<<B_ * R_ * (N1_ / 16), 256, LDS_TOTAL_BYTES, stream>>>(
      H1, H2, W2t, b2, w3, F2t, AR);

  k_mlp<<<B_ * (N1_ / 16), 256, 0, stream>>>(AR, Wa1t, ba1, Wa2t, ba2, (float*)d_out);
}
